// Qwen3DecoderLayer_6992206758016
// MI455X (gfx1250) — matrix operand order repaired and checked by reading, not compiled, not run on hardware
//
#include <hip/hip_runtime.h>
#include <hip/hip_bf16.h>

// ---------------------------------------------------------------------------
// Types for CDNA5 WMMA (wave32, 16x16x32 bf16 -> f32)
// ---------------------------------------------------------------------------
typedef __bf16 bf16_t;
typedef __attribute__((ext_vector_type(16))) __bf16 v16bf;
typedef __attribute__((ext_vector_type(8)))  __bf16 v8bf;
typedef __attribute__((ext_vector_type(8)))  float  v8f;

#define H_DIM  2048
#define NHQ    16
#define NKVH   8
#define HDIM   128
#define INTERD 8192
#define BATCH  2
#define SEQ    2048
#define TOK    (BATCH * SEQ)
#define EPSV   1e-6f
#define LN_THETA 13.815510557964274f   // ln(1e6)

__device__ __forceinline__ v8f wmma_bf16(v16bf a, v16bf b, v8f c) {
  return __builtin_amdgcn_wmma_f32_16x16x32_bf16(
      /*neg_a=*/false, a, /*neg_b=*/false, b,
      /*c_mod=*/(short)0, c, /*reuse_a=*/false, /*reuse_b=*/false);
}

__device__ __forceinline__ v8f zero_v8f() {
  v8f z = {0.f, 0.f, 0.f, 0.f, 0.f, 0.f, 0.f, 0.f};
  return z;
}

// CDNA5 async global->LDS copy (16B per lane), tracked with ASYNCcnt.
// LDS operand is the LDS byte offset = low 32 bits of the generic pointer
__device__ __forceinline__ void async_ld16(const bf16_t* g, bf16_t* l) {
  unsigned loff = (unsigned)(size_t)l;
  asm volatile("global_load_async_to_lds_b128 %0, %1, off"
               :
               : "v"(loff), "v"(g)
               : "memory");
}
__device__ __forceinline__ void wait_async0() {
  asm volatile("s_wait_asynccnt 0" ::: "memory");
}

// ---------------------------------------------------------------------------
// Weight convert + transpose: src f32 [K][N] -> dst bf16 [N][K]
// ---------------------------------------------------------------------------
__global__ __launch_bounds__(256)
void transpose_convert_kernel(const float* __restrict__ src,
                              bf16_t* __restrict__ dst, int K, int N) {
  __shared__ float tile[32][33];
  int nb = blockIdx.x * 32;
  int kb = blockIdx.y * 32;
  int tx = threadIdx.x & 31;
  int g  = threadIdx.x >> 5;       // 0..7
#pragma unroll
  for (int i = 0; i < 4; ++i) {
    int kk = g * 4 + i;
    tile[kk][tx] = src[(size_t)(kb + kk) * N + nb + tx];
  }
  __syncthreads();
#pragma unroll
  for (int i = 0; i < 4; ++i) {
    int nn = g * 4 + i;
    dst[(size_t)(nb + nn) * K + kb + tx] = (bf16_t)tile[tx][nn];
  }
}

// ---------------------------------------------------------------------------
// Fused (optional add) + RMSNorm. One block (256 thr) per token row of H_DIM.
// ---------------------------------------------------------------------------
template <bool HAS_ADD, bool WRITE_RES>
__global__ __launch_bounds__(256)
void add_rmsnorm_kernel(const float* __restrict__ a,
                        const float* __restrict__ b,
                        const float* __restrict__ w,
                        float* __restrict__ res_out,
                        bf16_t* __restrict__ xnorm) {
  int row = blockIdx.x;
  const float* ap = a + (size_t)row * H_DIM;
  const float* bp = HAS_ADD ? b + (size_t)row * H_DIM : nullptr;
  float vals[8];
  float ss = 0.f;
#pragma unroll
  for (int i = 0; i < 8; ++i) {
    int idx = threadIdx.x + i * 256;
    float v = ap[idx];
    if (HAS_ADD) v += bp[idx];
    vals[i] = v;
    ss += v * v;
  }
#pragma unroll
  for (int off = 16; off > 0; off >>= 1) ss += __shfl_xor(ss, off, 32);
  __shared__ float red[9];
  if ((threadIdx.x & 31) == 0) red[threadIdx.x >> 5] = ss;
  __syncthreads();
  if (threadIdx.x == 0) {
    float t = 0.f;
#pragma unroll
    for (int i = 0; i < 8; ++i) t += red[i];
    red[8] = rsqrtf(t / (float)H_DIM + EPSV);
  }
  __syncthreads();
  float rstd = red[8];
#pragma unroll
  for (int i = 0; i < 8; ++i) {
    int idx = threadIdx.x + i * 256;
    if (WRITE_RES) res_out[(size_t)row * H_DIM + idx] = vals[i];
    xnorm[(size_t)row * H_DIM + idx] = (bf16_t)(vals[i] * rstd * w[idx]);
  }
}

// ---------------------------------------------------------------------------
// bf16 GEMM: C[M,N] = A[M,K] * BT[N,K]^T (+bias[N]) (+resid[M,N])
// Block tile 128x128x32, 8 waves (2x4), wave tile 64x32 (4x2 WMMA tiles).
// Double-buffered LDS fed by async global->LDS (ASYNCcnt) copies.
// ---------------------------------------------------------------------------
#define BM 128
#define BN 128
#define BK 32
#define LDSS 40   // bf16 elems per LDS row (80B, keeps 16B alignment)

template <bool HAS_BIAS, bool HAS_RES, bool OUT_F32>
__global__ __launch_bounds__(256)
void gemm_bf16_kernel(const bf16_t* __restrict__ A,
                      const bf16_t* __restrict__ BT,
                      const float* __restrict__ bias,
                      const float* __restrict__ resid,
                      float* __restrict__ outf,
                      bf16_t* __restrict__ outb,
                      int M, int N, int K) {
  __shared__ alignas(16) bf16_t aT[2][BM * LDSS];
  __shared__ alignas(16) bf16_t bT[2][BN * LDSS];
  const int tid  = threadIdx.x;
  const int lane = tid & 31;
  const int wave = tid >> 5;
  const int wy = wave & 1;   // 2 wave rows
  const int wx = wave >> 1;  // 4 wave cols
  const int m0 = blockIdx.y * BM;
  const int n0 = blockIdx.x * BN;
  const int l15 = lane & 15;
  const int hf  = lane >> 4;

  v8f acc[4][2];
#pragma unroll
  for (int i = 0; i < 4; ++i)
#pragma unroll
    for (int j = 0; j < 2; ++j) acc[i][j] = zero_v8f();

  const int lr = tid >> 1;         // 0..127 tile row
  const int lc = (tid & 1) * 16;   // 0 or 16
  const bf16_t* agp = A  + (size_t)(m0 + lr) * K + lc;
  const bf16_t* bgp = BT + (size_t)(n0 + lr) * K + lc;

  auto issue_tile = [&](int buf, int k0) {
    async_ld16(agp + k0,     &aT[buf][lr * LDSS + lc]);
    async_ld16(agp + k0 + 8, &aT[buf][lr * LDSS + lc + 8]);
    async_ld16(bgp + k0,     &bT[buf][lr * LDSS + lc]);
    async_ld16(bgp + k0 + 8, &bT[buf][lr * LDSS + lc + 8]);
  };

  issue_tile(0, 0);
  const int nk = K / BK;
  for (int kk = 0; kk < nk; ++kk) {
    const int cur = kk & 1;
    wait_async0();        // this wave's async tile-k copies are in LDS
    __syncthreads();      // ... and everyone else's too
    if (kk + 1 < nk) issue_tile(cur ^ 1, (kk + 1) * BK);

    v16bf afrag[4];
    const int kbA = hf * 8;
#pragma unroll
    for (int i = 0; i < 4; ++i) {
      const bf16_t* p = &aT[cur][(wy * 64 + i * 16 + l15) * LDSS];
      *(v8bf*)&afrag[i]       = *(const v8bf*)(p + kbA);
      *((v8bf*)&afrag[i] + 1) = *(const v8bf*)(p + kbA + 16);
    }
    v16bf bfrag[2];
    const int kbB = hf * 8;
#pragma unroll
    for (int j = 0; j < 2; ++j) {
      const bf16_t* p = &bT[cur][(wx * 32 + j * 16 + l15) * LDSS + kbB];
      *(v8bf*)&bfrag[j]       = *(const v8bf*)(p);
      *((v8bf*)&bfrag[j] + 1) = *(const v8bf*)(p + 16);
    }
#pragma unroll
    for (int i = 0; i < 4; ++i)
#pragma unroll
      for (int j = 0; j < 2; ++j)
        acc[i][j] = wmma_bf16(afrag[i], bfrag[j], acc[i][j]);
    __syncthreads();      // frag reads of buf[cur] done before it is re-filled
  }

  // Epilogue. C layout: VGPR r holds M = r + 8*hf, N = l15 within each tile.
#pragma unroll
  for (int i = 0; i < 4; ++i)
#pragma unroll
    for (int j = 0; j < 2; ++j) {
      int col = n0 + wx * 32 + j * 16 + l15;
      float bv = HAS_BIAS ? bias[col] : 0.f;
#pragma unroll
      for (int r = 0; r < 8; ++r) {
        int row = m0 + wy * 64 + i * 16 + r + hf * 8;
        float v = acc[i][j][r] + bv;
        if (HAS_RES) v += resid[(size_t)row * N + col];
        if (OUT_F32) outf[(size_t)row * N + col] = v;
        else         outb[(size_t)row * N + col] = (bf16_t)v;
      }
    }
}

// ---------------------------------------------------------------------------
// RoPE in-place on bf16 [tok][heads][HDIM]; half-split rotation.
// ---------------------------------------------------------------------------
__global__ __launch_bounds__(64)
void rope_kernel(bf16_t* __restrict__ x, const int* __restrict__ positions,
                 int n_heads) {
  int t = blockIdx.x;
  int h = blockIdx.y;
  int i = threadIdx.x;  // 0..63
  float pos  = (float)positions[t];
  float freq = __expf(-(float)i * (LN_THETA / 64.0f));
  float ang  = pos * freq;
  float s, c;
  __sincosf(ang, &s, &c);
  bf16_t* p = x + ((size_t)t * n_heads + h) * HDIM;
  float x1 = (float)p[i];
  float x2 = (float)p[i + 64];
  p[i]      = (bf16_t)(x1 * c - x2 * s);
  p[i + 64] = (bf16_t)(x2 * c + x1 * s);
}

// ---------------------------------------------------------------------------
// V transpose: v [tok][NKVH][HDIM] -> vt [b][kvh][hd][SEQ]
// ---------------------------------------------------------------------------
__global__ __launch_bounds__(128)
void transpose_v_kernel(const bf16_t* __restrict__ v, bf16_t* __restrict__ vt) {
  int t = blockIdx.x;
  int h = blockIdx.y;
  int d = threadIdx.x;
  int b = t / SEQ, s = t % SEQ;
  vt[(((size_t)b * NKVH + h) * HDIM + d) * SEQ + s] =
      v[((size_t)t * NKVH + h) * HDIM + d];
}

// ---------------------------------------------------------------------------
// Causal flash attention (GQA). 4 waves/block, 16 q rows/wave, 32 kv/iter.
// grid: (BATCH*NHQ, SEQ/64), block 128.
// ---------------------------------------------------------------------------
__global__ __launch_bounds__(128)
void attn_kernel(const bf16_t* __restrict__ q, const bf16_t* __restrict__ k,
                 const bf16_t* __restrict__ vt, bf16_t* __restrict__ out) {
  const int bh  = blockIdx.x;
  const int b   = bh / NHQ;
  const int h   = bh % NHQ;
  const int kvh = h / (NHQ / NKVH);
  const int wave = threadIdx.x >> 5;
  const int lane = threadIdx.x & 31;
  const int l15  = lane & 15;
  const int hf   = lane >> 4;
  const int q0   = blockIdx.y * 64 + wave * 16;
  const float scale = 0.08838834764831845f;  // 1/sqrt(128)

  v16bf qf[4];
  {
    const bf16_t* qp = q + ((size_t)(b * SEQ + q0 + l15) * NHQ + h) * HDIM;
    const int kb = hf * 8;
#pragma unroll
    for (int c = 0; c < 4; ++c) {
      *(v8bf*)&qf[c]       = *(const v8bf*)(qp + c * 32 + kb);
      *((v8bf*)&qf[c] + 1) = *(const v8bf*)(qp + c * 32 + kb + 16);
    }
  }

  float mrow[8], lrow[8];
  v8f o[8];
#pragma unroll
  for (int j = 0; j < 8; ++j) { mrow[j] = -1e30f; lrow[j] = 0.f; }
#pragma unroll
  for (int nt = 0; nt < 8; ++nt) o[nt] = zero_v8f();

  __shared__ alignas(16) bf16_t plds[4][16][32];

  const int kv_hi = blockIdx.y * 64 + 64;  // uniform across waves (syncthreads!)
  for (int kv0 = 0; kv0 < kv_hi; kv0 += 32) {
    v8f s0 = zero_v8f(), s1 = zero_v8f();
#pragma unroll
    for (int c = 0; c < 4; ++c) {
      v16bf b0, b1;
      const bf16_t* p0 =
          k + ((size_t)(b * SEQ + kv0 + l15) * NKVH + kvh) * HDIM + c * 32 + hf * 8;
      const bf16_t* p1 =
          k + ((size_t)(b * SEQ + kv0 + 16 + l15) * NKVH + kvh) * HDIM + c * 32 + hf * 8;
      *(v8bf*)&b0 = *(const v8bf*)p0; *((v8bf*)&b0 + 1) = *(const v8bf*)(p0 + 16);
      *(v8bf*)&b1 = *(const v8bf*)p1; *((v8bf*)&b1 + 1) = *(const v8bf*)(p1 + 16);
      s0 = wmma_bf16(qf[c], b0, s0);
      s1 = wmma_bf16(qf[c], b1, s1);
    }
#pragma unroll
    for (int j = 0; j < 8; ++j) {
      int qrow = q0 + j + hf * 8;
      int c0 = kv0 + l15, c1 = kv0 + 16 + l15;
      float v0 = s0[j] * scale;
      float v1 = s1[j] * scale;
      if (c0 > qrow) v0 = -1e30f;
      if (c1 > qrow) v1 = -1e30f;
      float mx = fmaxf(v0, v1);
#pragma unroll
      for (int off = 8; off > 0; off >>= 1) mx = fmaxf(mx, __shfl_xor(mx, off, 32));
      float mnew = fmaxf(mrow[j], mx);
      float corr = __expf(mrow[j] - mnew);
      float p0v = __expf(v0 - mnew);
      float p1v = __expf(v1 - mnew);
      float rs = p0v + p1v;
#pragma unroll
      for (int off = 8; off > 0; off >>= 1) rs += __shfl_xor(rs, off, 32);
      lrow[j] = lrow[j] * corr + rs;
      mrow[j] = mnew;
#pragma unroll
      for (int nt = 0; nt < 8; ++nt) o[nt][j] *= corr;
      plds[wave][j + hf * 8][l15]      = (bf16_t)p0v;
      plds[wave][j + hf * 8][16 + l15] = (bf16_t)p1v;
    }
    __syncthreads();
    v16bf pf;
    {
      const bf16_t* pp = &plds[wave][l15][0];
      const int kb = hf * 8;
      *(v8bf*)&pf       = *(const v8bf*)(pp + kb);
      *((v8bf*)&pf + 1) = *(const v8bf*)(pp + kb + 16);
    }
    const bf16_t* vbase =
        vt + ((size_t)b * NKVH + kvh) * HDIM * SEQ + kv0 + hf * 8;
#pragma unroll
    for (int nt = 0; nt < 8; ++nt) {
      v16bf vf;
      const bf16_t* vp = vbase + (size_t)(nt * 16 + l15) * SEQ;
      *(v8bf*)&vf = *(const v8bf*)vp; *((v8bf*)&vf + 1) = *(const v8bf*)(vp + 16);
      o[nt] = wmma_bf16(pf, vf, o[nt]);
    }
    __syncthreads();
  }
#pragma unroll
  for (int nt = 0; nt < 8; ++nt)
#pragma unroll
    for (int j = 0; j < 8; ++j) {
      int row = q0 + j + hf * 8;
      out[((size_t)(b * SEQ + row)) * (NHQ * HDIM) + h * HDIM + nt * 16 + l15] =
          (bf16_t)(o[nt][j] / lrow[j]);
    }
}

// ---------------------------------------------------------------------------
// SwiGLU: out = silu(gate) * up  (bf16, elementwise; out may alias gate)
// ---------------------------------------------------------------------------
__global__ __launch_bounds__(256)
void silu_mul_kernel(const bf16_t* __restrict__ gate,
                     const bf16_t* __restrict__ up,
                     bf16_t* __restrict__ out, size_t n) {
  size_t i = (size_t)blockIdx.x * blockDim.x + threadIdx.x;
  if (i < n) {
    float g = (float)gate[i];
    float u = (float)up[i];
    float s = g / (1.f + __expf(-g));
    out[i] = (bf16_t)(s * u);
  }
}

// ---------------------------------------------------------------------------
// Host orchestration
// ---------------------------------------------------------------------------
extern "C" void kernel_launch(void* const* d_in, const int* in_sizes, int n_in,
                              void* d_out, int out_size, void* d_ws, size_t ws_size,
                              hipStream_t stream) {
  const int*   positions = (const int*)d_in[0];
  const float* hidden    = (const float*)d_in[1];
  const float* residual  = (const float*)d_in[2];
  const float* ln1_w     = (const float*)d_in[3];
  const float* ln2_w     = (const float*)d_in[4];
  const float* wq        = (const float*)d_in[5];
  const float* bq        = (const float*)d_in[6];
  const float* wk        = (const float*)d_in[7];
  const float* bk        = (const float*)d_in[8];
  const float* wv        = (const float*)d_in[9];
  const float* bv        = (const float*)d_in[10];
  const float* wo        = (const float*)d_in[11];
  const float* wgu       = (const float*)d_in[12];
  const float* wdn       = (const float*)d_in[13];
  float* out = (float*)d_out;                 // [h | res], each TOK*H_DIM f32
  float* out_h   = out;
  float* out_res = out + (size_t)TOK * H_DIM;

  const int QD  = NHQ * HDIM;    // 2048
  const int KVD = NKVH * HDIM;   // 1024

  char* ws = (char*)d_ws;
  size_t off = 0;
  auto alloc = [&](size_t bytes) -> void* {
    void* p = ws + off;
    off += (bytes + 255) & ~(size_t)255;
    return p;
  };
  bf16_t* wqT  = (bf16_t*)alloc((size_t)QD * H_DIM * 2);
  bf16_t* wkT  = (bf16_t*)alloc((size_t)KVD * H_DIM * 2);
  bf16_t* wvT  = (bf16_t*)alloc((size_t)KVD * H_DIM * 2);
  bf16_t* woT  = (bf16_t*)alloc((size_t)H_DIM * QD * 2);
  bf16_t* wguT = (bf16_t*)alloc((size_t)(2 * INTERD) * H_DIM * 2);
  bf16_t* wdnT = (bf16_t*)alloc((size_t)H_DIM * INTERD * 2);
  float*  res1 = (float*)alloc((size_t)TOK * H_DIM * 4);
  bf16_t* xnorm = (bf16_t*)alloc((size_t)TOK * H_DIM * 2);
  bf16_t* qb   = (bf16_t*)alloc((size_t)TOK * QD * 2);
  bf16_t* kb_  = (bf16_t*)alloc((size_t)TOK * KVD * 2);
  bf16_t* vb   = (bf16_t*)alloc((size_t)TOK * KVD * 2);
  bf16_t* vtb  = (bf16_t*)alloc((size_t)TOK * KVD * 2);
  bf16_t* attnb = (bf16_t*)alloc((size_t)TOK * QD * 2);
  bf16_t* gate = (bf16_t*)alloc((size_t)TOK * INTERD * 2);
  bf16_t* up   = (bf16_t*)alloc((size_t)TOK * INTERD * 2);

  // 1) weight convert+transpose (bf16, [N][K])
  transpose_convert_kernel<<<dim3(QD / 32, H_DIM / 32), 256, 0, stream>>>(wq, wqT, H_DIM, QD);
  transpose_convert_kernel<<<dim3(KVD / 32, H_DIM / 32), 256, 0, stream>>>(wk, wkT, H_DIM, KVD);
  transpose_convert_kernel<<<dim3(KVD / 32, H_DIM / 32), 256, 0, stream>>>(wv, wvT, H_DIM, KVD);
  transpose_convert_kernel<<<dim3(H_DIM / 32, QD / 32), 256, 0, stream>>>(wo, woT, QD, H_DIM);
  transpose_convert_kernel<<<dim3((2 * INTERD) / 32, H_DIM / 32), 256, 0, stream>>>(wgu, wguT, H_DIM, 2 * INTERD);
  transpose_convert_kernel<<<dim3(H_DIM / 32, INTERD / 32), 256, 0, stream>>>(wdn, wdnT, INTERD, H_DIM);

  // 2) res1 = hidden + residual; xnorm = rmsnorm(res1)*ln1_w (bf16)
  add_rmsnorm_kernel<true, true><<<TOK, 256, 0, stream>>>(hidden, residual, ln1_w, res1, xnorm);

  // 3) QKV projections (bias fused, bf16 out)
  gemm_bf16_kernel<true, false, false><<<dim3(QD / BN, TOK / BM), 256, 0, stream>>>(
      xnorm, wqT, bq, nullptr, nullptr, qb, TOK, QD, H_DIM);
  gemm_bf16_kernel<true, false, false><<<dim3(KVD / BN, TOK / BM), 256, 0, stream>>>(
      xnorm, wkT, bk, nullptr, nullptr, kb_, TOK, KVD, H_DIM);
  gemm_bf16_kernel<true, false, false><<<dim3(KVD / BN, TOK / BM), 256, 0, stream>>>(
      xnorm, wvT, bv, nullptr, nullptr, vb, TOK, KVD, H_DIM);

  // 4) RoPE on q and k
  rope_kernel<<<dim3(TOK, NHQ), 64, 0, stream>>>(qb, positions, NHQ);
  rope_kernel<<<dim3(TOK, NKVH), 64, 0, stream>>>(kb_, positions, NKVH);

  // 5) V transpose for WMMA B-fragments
  transpose_v_kernel<<<dim3(TOK, NKVH), HDIM, 0, stream>>>(vb, vtb);

  // 6) causal GQA flash attention
  attn_kernel<<<dim3(BATCH * NHQ, SEQ / 64), 128, 0, stream>>>(qb, kb_, vtb, attnb);

  // 7) O projection with fused residual add -> res2 (f32, into d_out)
  gemm_bf16_kernel<false, true, true><<<dim3(H_DIM / BN, TOK / BM), 256, 0, stream>>>(
      attnb, woT, nullptr, res1, out_res, nullptr, TOK, H_DIM, QD);

  // 8) post-attention RMSNorm
  add_rmsnorm_kernel<false, false><<<TOK, 256, 0, stream>>>(out_res, nullptr, ln2_w, nullptr, xnorm);

  // 9) gate / up projections (bf16 out)
  gemm_bf16_kernel<false, false, false><<<dim3(INTERD / BN, TOK / BM), 256, 0, stream>>>(
      xnorm, wguT, nullptr, nullptr, nullptr, gate, TOK, INTERD, H_DIM);
  gemm_bf16_kernel<false, false, false><<<dim3(INTERD / BN, TOK / BM), 256, 0, stream>>>(
      xnorm, wguT + (size_t)INTERD * H_DIM, nullptr, nullptr, nullptr, up,
      TOK, INTERD, H_DIM);

  // 10) SwiGLU (in place into gate)
  {
    size_t n = (size_t)TOK * INTERD;
    silu_mul_kernel<<<(unsigned)((n + 255) / 256), 256, 0, stream>>>(gate, up, gate, n);
  }

  // 11) down projection -> h (f32, into d_out)
  gemm_bf16_kernel<false, false, true><<<dim3(H_DIM / BN, TOK / BM), 256, 0, stream>>>(
      gate, wdnT, nullptr, nullptr, out_h, nullptr, TOK, H_DIM, INTERD);
}
